// CausalFNetBlock_47519518163599
// MI455X (gfx1250) — matrix operand order repaired and checked by reading, not compiled, not run on hardware
//
#include <hip/hip_runtime.h>

// ---------------------------------------------------------------------------
// CausalFNetBlock for MI455X (gfx1250): bf16 WMMA GEMMs with async-to-LDS
// double-buffered tiles + exact IIR scan replacing the FFT convolution
// (filter exp(-0.1*t) => first-order IIR recurrence).
// ---------------------------------------------------------------------------

#define B_SZ 4
#define T_SZ 4096
#define D_SZ 1024
#define NHEADS 8
#define MLP_H 2048
#define EPSV 1e-6f

typedef __attribute__((ext_vector_type(16))) __bf16 v16bf;
typedef __attribute__((ext_vector_type(8)))  __bf16 v8bf;
typedef __attribute__((ext_vector_type(8)))  float  v8f;

// --- async global->LDS copy (16B per lane), ASYNCcnt-tracked ---------------
#if defined(__has_builtin)
#if __has_builtin(__builtin_amdgcn_global_load_async_to_lds_b128)
#define HAVE_ASYNC_BUILTIN 1
#endif
#if __has_builtin(__builtin_amdgcn_s_wait_asynccnt)
#define HAVE_ASYNCWAIT_BUILTIN 1
#endif
#endif

typedef int v4i_gcc __attribute__((vector_size(16)));
typedef __attribute__((address_space(1))) v4i_gcc* as1_v4i;
typedef __attribute__((address_space(3))) v4i_gcc* as3_v4i;

__device__ __forceinline__ void async_cp16(const __bf16* gsrc, __bf16* ldst) {
#if defined(HAVE_ASYNC_BUILTIN)
    // Generic global address == AS1 address; low 32 bits of a generic LDS
    // address == LDS byte offset (ISA 10.2 aperture mapping).
    __builtin_amdgcn_global_load_async_to_lds_b128(
        (as1_v4i)(size_t)gsrc,
        (as3_v4i)(unsigned)(size_t)ldst, 0, 0);
#else
    asm volatile("global_load_async_to_lds_b128 %0, %1, off"
                 :
                 : "v"((unsigned)(size_t)ldst), "v"((unsigned long long)(size_t)gsrc)
                 : "memory");
#endif
}

__device__ __forceinline__ void async_wait0() {
#if defined(HAVE_ASYNCWAIT_BUILTIN)
    __builtin_amdgcn_s_wait_asynccnt(0);
#else
    asm volatile("s_wait_asynccnt 0" ::: "memory");
#endif
}

// ---------------------------------------------------------------------------
// Fake-int8 quantize each row of W (N x K), store bf16 for WMMA consumption.
// ---------------------------------------------------------------------------
__global__ __launch_bounds__(256)
void quantize_bf16_kernel(const float* __restrict__ w, __bf16* __restrict__ wq, int K) {
    const int row = blockIdx.x;
    const float* wr = w + (size_t)row * K;
    __bf16* wqr = wq + (size_t)row * K;
    __shared__ float red[9];
    float m = 0.f;
    for (int i = threadIdx.x; i < K; i += 256) m = fmaxf(m, fabsf(wr[i]));
    for (int o = 16; o > 0; o >>= 1) m = fmaxf(m, __shfl_down(m, o, 32));
    if ((threadIdx.x & 31) == 0) red[threadIdx.x >> 5] = m;
    __syncthreads();
    if (threadIdx.x == 0) {
        float t = red[0];
        for (int i = 1; i < 8; ++i) t = fmaxf(t, red[i]);
        red[8] = fmaxf(t * (1.0f / 127.0f), 1e-8f);
    }
    __syncthreads();
    const float scale = red[8];
    const float inv = 1.0f / scale;
    for (int i = threadIdx.x; i < K; i += 256)
        wqr[i] = (__bf16)(rintf(wr[i] * inv) * scale);
}

// ---------------------------------------------------------------------------
// RMSNorm one row of (rows x D), output bf16.
// ---------------------------------------------------------------------------
__global__ __launch_bounds__(256)
void rmsnorm_bf16_kernel(const float* __restrict__ x, const float* __restrict__ w,
                         __bf16* __restrict__ y, int D) {
    const int row = blockIdx.x;
    const float* xr = x + (size_t)row * D;
    __bf16* yr = y + (size_t)row * D;
    __shared__ float red[9];
    float ss = 0.f;
    for (int i = threadIdx.x; i < D; i += 256) { float v = xr[i]; ss = fmaf(v, v, ss); }
    for (int o = 16; o > 0; o >>= 1) ss += __shfl_down(ss, o, 32);
    if ((threadIdx.x & 31) == 0) red[threadIdx.x >> 5] = ss;
    __syncthreads();
    if (threadIdx.x == 0) {
        float t = 0.f;
        for (int i = 0; i < 8; ++i) t += red[i];
        red[8] = rsqrtf(t / (float)D + EPSV);
    }
    __syncthreads();
    const float rs = red[8];
    for (int i = threadIdx.x; i < D; i += 256)
        yr[i] = (__bf16)(xr[i] * rs * w[i]);
}

// ---------------------------------------------------------------------------
// IIR scan (y[t] = s[t] + a*y[t-1]) fused with silu-gating.
// Chunk 256 + warm-up 256: a^256 = exp(-25.6) ~ 7.5e-12 => exact to f32.
// ---------------------------------------------------------------------------
__global__ __launch_bounds__(256)
void scan_gate_kernel(const float* __restrict__ qkv, const float* __restrict__ filt,
                      __bf16* __restrict__ u) {
    const int d  = blockIdx.x * 256 + threadIdx.x;
    const int b  = blockIdx.z;
    const int c0 = blockIdx.y * 256;
    const float a = filt[NHEADS];   // filt[t=1,h=0] = exp(-0.1); filt[0]=1
    float y = 0.f;
    int t0 = c0 - 256; if (t0 < 0) t0 = 0;
    for (int t = t0; t < c0 + 256; ++t) {
        const size_t base = ((size_t)b * T_SZ + t) * (2 * D_SZ);
        y = fmaf(a, y, qkv[base + d]);
        if (t >= c0) {
            const float g = qkv[base + D_SZ + d];
            const float sil = g / (1.0f + __expf(-g));
            u[((size_t)b * T_SZ + t) * D_SZ + d] = (__bf16)(y * sil);
        }
    }
}

// ---------------------------------------------------------------------------
// WMMA GEMM: C[M,N] = A[M,K] (bf16) * W[N,K]^T (bf16), f32 accumulate.
// N, K are compile-time so all strides/offsets fold to immediates.
// Block = 256 threads = 8 waves (4M x 2N). Block tile 128M x 128N, k-step 32,
// double-buffered in LDS via GLOBAL_LOAD_ASYNC_TO_LDS_B128 (ASYNCcnt).
// Wave tile = 32M x 64N: 2 A frags x 4 B frags -> 8 v_wmma per k-step.
// LDS layout: tile[row][kk] row-major, row stride 32 elements (64B).
// MODE: 0 = store f32; 1 = +residual, store f32; 2 = exact GELU, store bf16.
// ---------------------------------------------------------------------------
template <int MODE, int N, int K>
__global__ __launch_bounds__(256)
void gemm_bf16_wmma_kernel(const __bf16* __restrict__ A, const __bf16* __restrict__ W,
                           const float* __restrict__ resid, void* __restrict__ out) {
    __shared__ __bf16 As[2][128 * 32];
    __shared__ __bf16 Bs[2][128 * 32];

    const int tid   = threadIdx.x;
    const int wave  = tid >> 5;
    const int lane  = tid & 31;
    const int l16   = lane & 15;
    const int hi    = lane >> 4;
    const int waveM = wave >> 1;          // 0..3
    const int waveN = wave & 1;           // 0..1
    const int tileM0 = blockIdx.y * 128;
    const int tileN0 = blockIdx.x * 128;

    // copy-chunk mapping: 512 chunks of 16B per tile; thread owns 2 chunks
    const int c0r = (tid + 0)   >> 2, c0c = (tid + 0)   & 3;
    const int c1r = (tid + 256) >> 2, c1c = (tid + 256) & 3;

    const __bf16* gA0 = A + (size_t)(tileM0 + c0r) * K + c0c * 8;
    const __bf16* gA1 = A + (size_t)(tileM0 + c1r) * K + c1c * 8;
    const __bf16* gB0 = W + (size_t)(tileN0 + c0r) * K + c0c * 8;
    const __bf16* gB1 = W + (size_t)(tileN0 + c1r) * K + c1c * 8;
    __bf16* lA0 = &As[0][c0r * 32 + c0c * 8];
    __bf16* lA1 = &As[0][c1r * 32 + c1c * 8];
    __bf16* lB0 = &Bs[0][c0r * 32 + c0c * 8];
    __bf16* lB1 = &Bs[0][c1r * 32 + c1c * 8];
    constexpr int LBUF = 128 * 32;        // elements per LDS buffer

    constexpr int nk = K / 32;

    // --- prologue: fill buffer 0 ---
    async_cp16(gA0, lA0);
    async_cp16(gA1, lA1);
    async_cp16(gB0, lB0);
    async_cp16(gB1, lB1);
    async_wait0();
    __syncthreads();

    v8f acc[2][4];
    #pragma unroll
    for (int s = 0; s < 2; ++s)
        #pragma unroll
        for (int j = 0; j < 4; ++j)
            #pragma unroll
            for (int r = 0; r < 8; ++r) acc[s][j][r] = 0.f;

    #pragma unroll 2
    for (int kt = 0; kt < nk; ++kt) {
        const int buf = kt & 1;           // static under unroll-by-2
        if (kt + 1 < nk) {
            const int nb = buf ^ 1;
            const int kk = (kt + 1) * 32;
            async_cp16(gA0 + kk, lA0 + nb * LBUF);
            async_cp16(gA1 + kk, lA1 + nb * LBUF);
            async_cp16(gB0 + kk, lB0 + nb * LBUF);
            async_cp16(gB1 + kk, lB1 + nb * LBUF);
        }

        // A fragments: rows waveM*32 + s*16 + l16; bytes [hi*32, +32)
        v16bf afrag[2];
        #pragma unroll
        for (int s = 0; s < 2; ++s) {
            const int row = waveM * 32 + s * 16 + l16;
            const v8bf a0 = *(const v8bf*)&As[buf][row * 32 + hi * 16];
            const v8bf a1 = *(const v8bf*)&As[buf][row * 32 + hi * 16 + 8];
            #pragma unroll
            for (int i = 0; i < 8; ++i) { afrag[s][i] = a0[i]; afrag[s][8 + i] = a1[i]; }
        }

        #pragma unroll
        for (int j = 0; j < 4; ++j) {
            const int row = waveN * 64 + j * 16 + l16;
            const v16bf bfrag = *(const v16bf*)&Bs[buf][row * 32 + hi * 16];
            #pragma unroll
            for (int s = 0; s < 2; ++s) {
                acc[s][j] = __builtin_amdgcn_wmma_f32_16x16x32_bf16(
                    false, afrag[s], false, bfrag, (short)0, acc[s][j], false, false);
            }
        }

        if (kt + 1 < nk) async_wait0();
        __syncthreads();
    }

    // epilogue: C 16x16 layout -> vgpr r: row hi*8+r, col l16
    #pragma unroll
    for (int s = 0; s < 2; ++s) {
        const int row0 = tileM0 + waveM * 32 + s * 16 + hi * 8;
        #pragma unroll
        for (int j = 0; j < 4; ++j) {
            const int col = tileN0 + waveN * 64 + j * 16 + l16;
            const size_t base = (size_t)row0 * N + col;
            #pragma unroll
            for (int r = 0; r < 8; ++r) {
                const size_t idx = base + (size_t)r * N;
                float v = acc[s][j][r];
                if (MODE == 2) {
                    const float g = 0.5f * v * (1.0f + erff(v * 0.70710678f));
                    ((__bf16*)out)[idx] = (__bf16)g;
                } else {
                    if (MODE == 1) v += resid[idx];
                    ((float*)out)[idx] = v;
                }
            }
        }
    }
}

// ---------------------------------------------------------------------------
// Launcher
// ---------------------------------------------------------------------------
extern "C" void kernel_launch(void* const* d_in, const int* in_sizes, int n_in,
                              void* d_out, int out_size, void* d_ws, size_t ws_size,
                              hipStream_t stream) {
    (void)in_sizes; (void)n_in; (void)out_size; (void)ws_size;
    const float* x    = (const float*)d_in[0];
    const float* n1w  = (const float*)d_in[1];
    const float* n2w  = (const float*)d_in[2];
    const float* w_in = (const float*)d_in[3];
    const float* w_o  = (const float*)d_in[4];
    const float* w_mi = (const float*)d_in[5];
    const float* w_mo = (const float*)d_in[6];
    const float* filt = (const float*)d_in[7];

    const int M = B_SZ * T_SZ;              // 16384 rows
    const size_t MB = 1ull << 20;
    char* ws = (char*)d_ws;
    float*  qkv  = (float*)(ws + 0);          // 32M f32  = 128MB
    __bf16* gelu = (__bf16*)(ws + 0);         // reuse after qkv is dead
    __bf16* u    = (__bf16*)(ws + 128 * MB);  // 16M bf16 = 32MB
    float*  x2   = (float*)(ws + 160 * MB);   // 16M f32  = 64MB
    __bf16* hn   = (__bf16*)(ws + 224 * MB);  // 16M bf16 = 32MB (hnorm, then h2)
    __bf16* wqin = (__bf16*)(ws + 256 * MB);  // 4MB
    __bf16* wqo  = (__bf16*)(ws + 260 * MB);  // 2MB
    __bf16* wqmi = (__bf16*)(ws + 262 * MB);  // 4MB
    __bf16* wqmo = (__bf16*)(ws + 266 * MB);  // 4MB (total 270MB)

    // 1) fake-int8 quantize all weights -> bf16
    quantize_bf16_kernel<<<2 * D_SZ, 256, 0, stream>>>(w_in, wqin, D_SZ);
    quantize_bf16_kernel<<<D_SZ,     256, 0, stream>>>(w_o,  wqo,  D_SZ);
    quantize_bf16_kernel<<<MLP_H,    256, 0, stream>>>(w_mi, wqmi, D_SZ);
    quantize_bf16_kernel<<<D_SZ,     256, 0, stream>>>(w_mo, wqmo, MLP_H);

    // 2) h = rmsnorm(x, norm1) -> bf16
    rmsnorm_bf16_kernel<<<M, 256, 0, stream>>>(x, n1w, hn, D_SZ);

    // 3) qkv = h @ Wq_in^T   (M x 2D)
    gemm_bf16_wmma_kernel<0, 2 * D_SZ, D_SZ>
        <<<dim3((2 * D_SZ) / 128, M / 128), 256, 0, stream>>>(hn, wqin, nullptr, qkv);

    // 4) IIR scan + silu gating -> u (bf16)
    scan_gate_kernel<<<dim3(D_SZ / 256, T_SZ / 256, B_SZ), 256, 0, stream>>>(qkv, filt, u);

    // 5) x2 = x + u @ Wq_o^T
    gemm_bf16_wmma_kernel<1, D_SZ, D_SZ>
        <<<dim3(D_SZ / 128, M / 128), 256, 0, stream>>>(u, wqo, x, x2);

    // 6) h2 = rmsnorm(x2, norm2) -> bf16 (reuse hn)
    rmsnorm_bf16_kernel<<<M, 256, 0, stream>>>(x2, n2w, hn, D_SZ);

    // 7) m = gelu(h2 @ Wq_mi^T) -> bf16 (reuse qkv buffer region)
    gemm_bf16_wmma_kernel<2, MLP_H, D_SZ>
        <<<dim3(MLP_H / 128, M / 128), 256, 0, stream>>>(hn, wqmi, nullptr, gelu);

    // 8) out = x2 + m @ Wq_mo^T  (f32 output)
    gemm_bf16_wmma_kernel<1, D_SZ, MLP_H>
        <<<dim3(D_SZ / 128, M / 128), 256, 0, stream>>>(gelu, wqmo, x2, (float*)d_out);
}
